// DCGRUEncoder_10471130268000
// MI455X (gfx1250) — matrix operand order repaired and checked by reading, not compiled, not run on hardware
//
#include <hip/hip_runtime.h>

// ---------------------------------------------------------------------------
// DCGRU encoder for MI455X (gfx1250, wave32, WMMA).
//  - GEMMs run on v_wmma_f32_16x16x32_bf16 with f32 accumulation.
//  - A (activations) converted f32->bf16 once per diff-conv; fragment layout
//    matches two contiguous global_load_b128 per lane per k-step.
//  - B (weights) pre-packed into WMMA fragment order: one 32B coalesced read
//    per lane per (k-step, n-tile). No VALU packing in the hot loop.
//  - Each wave computes a 16x(16*NTW) strip: NTW WMMAs per k iteration.
//  - Graph scatters (diff-conv hops, GAT softmax) -> f32 global atomics.
// All launches on `stream`; no allocation/sync (graph-capture safe).
// ---------------------------------------------------------------------------

#define TT 96
#define NS 2000
#define NI 20000
#define NE 10000
#define ES 32000
#define EI 18000
#define EE 18000
#define HIDF 128
#define MEASF 16
#define STATF 8
#define NWPF 32
#define HEADS 4
#define HD 8

typedef __attribute__((ext_vector_type(16))) __bf16 bf16x16;
typedef __attribute__((ext_vector_type(8)))  float  floatx8;

union BF16Vec { bf16x16 v; unsigned short u[16]; uint4 q[2]; };

__device__ __forceinline__ unsigned short f2bf(float f) {
  unsigned int u = __float_as_uint(f);
  unsigned int r = u + 0x7FFFu + ((u >> 16) & 1u);   // round-to-nearest-even
  return (unsigned short)(r >> 16);
}

__device__ __forceinline__ void atomicMaxF(float* addr, float val) {
  unsigned int* ua = (unsigned int*)addr;
  unsigned int cur = *ua;
  while (__uint_as_float(cur) < val) {
    unsigned int prev = atomicCAS(ua, cur, __float_as_uint(val));
    if (prev == cur) break;
    cur = prev;
  }
}

// ---------------------------------------------------------------------------
__global__ void k_fill(float* p, int n, float v) {
  int i = blockIdx.x * blockDim.x + threadIdx.x;
  if (i < n) p[i] = v;
}

// f32 -> bf16 bits, elementwise
__global__ void k_cvt_bf16(const float* __restrict__ in, unsigned short* __restrict__ o, int n) {
  int i = blockIdx.x * blockDim.x + threadIdx.x;
  if (i < n) o[i] = f2bf(in[i]);
}

// Pack W (K x N f32, zero-padded to Kpad) into WMMA B-fragment order:
// out[(((kt*tN)+tn)*32 + lane)*16 + e] = bf16(W[kt*32 + (e>>3)*16 + (lane>>4)*8 + (e&7), tn*16 + (lane&15)])
// so the GEMM reads each lane's fragment as two contiguous 16B loads.
__global__ void k_pack_b(const float* __restrict__ W, unsigned short* __restrict__ o,
                         int K, int Kpad, int N) {
  int i = blockIdx.x * blockDim.x + threadIdx.x;
  if (i >= Kpad * N) return;
  int idx = i;
  int e    = idx & 15;  idx >>= 4;
  int lane = idx & 31;  idx >>= 5;
  int tN   = N >> 4;
  int tn   = idx % tN;
  int kt   = idx / tN;
  int r = lane & 15, hi = lane >> 4;
  int kk  = kt * 32 + (e >> 3) * 16 + hi * 8 + (e & 7);
  int col = tn * 16 + r;
  o[i] = (kk < K) ? f2bf(W[kk * N + col]) : (unsigned short)0;
}

// ---------------------------------------------------------------------------
// GEMM: C[M,N] = act(A[M,K](bf16) * B(pre-packed bf16) + bias[N])
// One wave -> 16 x (16*NTW) strip of C. Requires M%16==0, N%(16*NTW)==0,
// K%32==0 (A pre-padded with zeros; B pad rows are zero -> exact).
// act: 0=none 1=sigmoid 2=tanh
// ---------------------------------------------------------------------------
template <int NTW>
__global__ void k_gemm_wmma(const unsigned short* __restrict__ A,
                            const unsigned short* __restrict__ Bp,
                            const float* __restrict__ bias, float* __restrict__ C,
                            int M, int N, int K, int act) {
  int wid  = blockIdx.x * (blockDim.x >> 5) + (threadIdx.x >> 5);
  int lane = threadIdx.x & 31;
  int tM = M >> 4, tN = N >> 4, tG = tN / NTW;
  if (wid >= tM * tG) return;
  int tm = wid / tG, tg = wid % tG;
  int r = lane & 15, hi = lane >> 4;

  floatx8 acc[NTW];
#pragma unroll
  for (int n = 0; n < NTW; ++n) acc[n] = (floatx8){0.f,0.f,0.f,0.f,0.f,0.f,0.f,0.f};

  const unsigned short* Arow = A + (size_t)(tm * 16 + r) * K;
  const unsigned short* Bg   = Bp + ((size_t)tg * NTW * 32 + lane) * 16;
  const size_t bKt = (size_t)tN * 32 * 16;   // packed elems per k-step across all n-tiles

  for (int k = 0; k < K; k += 32) {
    __builtin_prefetch((const void*)(Arow + k + 64), 0, 1);  // global_prefetch
    BF16Vec a;
    a.q[0] = *(const uint4*)(Arow + k + hi * 8);        // K = k + hi*8 .. +7
    a.q[1] = *(const uint4*)(Arow + k + 16 + hi * 8);   // K = k+16+hi*8 .. +7
    const unsigned short* bp = Bg + (size_t)(k >> 5) * bKt;
#pragma unroll
    for (int n = 0; n < NTW; ++n) {
      BF16Vec b;
      b.q[0] = *(const uint4*)(bp + (size_t)n * 32 * 16);
      b.q[1] = *(const uint4*)(bp + (size_t)n * 32 * 16 + 8);
      acc[n] = __builtin_amdgcn_wmma_f32_16x16x32_bf16(false, a.v, false, b.v,
                                                       (short)0, acc[n], false, false);
    }
  }

#pragma unroll
  for (int n = 0; n < NTW; ++n) {
    int col = (tg * NTW + n) * 16 + r;
    float bv = bias[col];
#pragma unroll
    for (int j = 0; j < 8; ++j) {
      int mo = tm * 16 + j + hi * 8;
      float x = acc[n][j] + bv;
      if (act == 1)      x = 1.f / (1.f + __expf(-x));
      else if (act == 2) x = tanhf(x);
      C[(size_t)mo * N + col] = x;
    }
  }
}

// ---------------------------------------------------------------------------
// One diffusion-conv hop: dc[dst, out_off+f] += w * dc[src, in_off+f]
// ---------------------------------------------------------------------------
__global__ void k_spmv_hop(const int* __restrict__ src, const int* __restrict__ dst,
                           const float* __restrict__ w, int E,
                           float* dc, int stride, int in_off, int out_off, int nf) {
  int i = blockIdx.x * blockDim.x + threadIdx.x;
  if (i >= E * nf) return;
  int e = i / nf, f = i % nf;
  atomicAdd(&dc[(size_t)dst[e] * stride + out_off + f],
            w[e] * dc[(size_t)src[e] * stride + in_off + f]);
}

// cell0 diff-conv input: [x(56) | (g? r*h : h)(128) | zeros to 576]
__global__ void k_build_dc0(const float* __restrict__ meas, const float* __restrict__ nwp,
                            const float* __restrict__ stat, const float* __restrict__ h,
                            const float* __restrict__ g, float* __restrict__ dc) {
  int i = blockIdx.x * blockDim.x + threadIdx.x;
  if (i >= NS * 576) return;
  int node = i / 576, f = i % 576;
  float v = 0.f;
  if (f < 16)       v = meas[node * MEASF + f];
  else if (f < 48)  v = nwp[node * NWPF + (f - 16)];
  else if (f < 56)  v = stat[node * STATF + (f - 48)];
  else if (f < 184) {
    float hv = h[node * HIDF + (f - 56)];
    v = g ? g[node * 256 + (f - 56)] * hv : hv;
  }
  dc[i] = v;
}

// cell1 diff-conv input: [h0(128) | (g? r*h1 : h1)(128) | zeros to 768]
__global__ void k_build_dc1(const float* __restrict__ h0, const float* __restrict__ h1,
                            const float* __restrict__ g, float* __restrict__ dc) {
  int i = blockIdx.x * blockDim.x + threadIdx.x;
  if (i >= NS * 768) return;
  int node = i / 768, f = i % 768;
  float v = 0.f;
  if (f < 128)      v = h0[node * HIDF + f];
  else if (f < 256) {
    float hv = h1[node * HIDF + (f - 128)];
    v = g ? g[node * 256 + (f - 128)] * hv : hv;
  }
  dc[i] = v;
}

// h = u*h + (1-u)*c  with u = g[:,128:]
__global__ void k_update(float* __restrict__ h, const float* __restrict__ g,
                         const float* __restrict__ c) {
  int i = blockIdx.x * blockDim.x + threadIdx.x;
  if (i >= NS * HIDF) return;
  int node = i / HIDF, f = i % HIDF;
  float u = g[node * 256 + 128 + f];
  h[i] = u * h[i] + (1.f - u) * c[i];
}

// ---------------------------------------------------------------------------
// NWP attention. wfa[h] = Wf_block[h] @ a[h] (3-vec); wea likewise.
// ---------------------------------------------------------------------------
__global__ void k_nwp_prep(const float* __restrict__ Wf, const float* __restrict__ We,
                           const float* __restrict__ a,
                           float* __restrict__ wfa, float* __restrict__ wea) {
  int i = threadIdx.x;
  if (i >= HEADS * 3) return;
  int h = i / 3, c = i % 3;
  float s1 = 0.f, s2 = 0.f;
  for (int d = 0; d < HD; ++d) {
    float ah = a[h * HD + d];
    s1 += Wf[c * (HEADS * HD) + h * HD + d] * ah;
    s2 += We[c * (HEADS * HD) + h * HD + d] * ah;
  }
  wfa[i] = s1; wea[i] = s2;
}

__global__ void k_nwp_logit(const float* __restrict__ seq, const int* __restrict__ src,
                            const int* __restrict__ dst, const float* __restrict__ ea,
                            const float* __restrict__ wfa, const float* __restrict__ wea,
                            float* __restrict__ logit, float* __restrict__ maxb,
                            int E, int NN) {
  int i = blockIdx.x * blockDim.x + threadIdx.x;
  if (i >= TT * E) return;
  int t = i / E, e = i % E;
  const float* fp = seq + ((size_t)t * NN + src[e]) * 3;
  const float* av = ea + (size_t)e * 3;
  int d = dst[e];
  float f0 = fp[0], f1 = fp[1], f2 = fp[2];
  float a0 = av[0], a1 = av[1], a2 = av[2];
#pragma unroll
  for (int h = 0; h < HEADS; ++h) {
    float l = f0 * wfa[h*3] + f1 * wfa[h*3+1] + f2 * wfa[h*3+2]
            + a0 * wea[h*3] + a1 * wea[h*3+1] + a2 * wea[h*3+2];
    l = (l > 0.f) ? l : 0.2f * l;                 // leaky_relu(0.2)
    logit[(size_t)i * HEADS + h] = l;
    atomicMaxF(&maxb[((size_t)t * NS + d) * HEADS + h], l);
  }
}

__global__ void k_nwp_exp(const int* __restrict__ dst, float* __restrict__ logit,
                          const float* __restrict__ maxb, float* __restrict__ sumb, int E) {
  int i = blockIdx.x * blockDim.x + threadIdx.x;
  if (i >= TT * E) return;
  int t = i / E, e = i % E;
  int d = dst[e];
#pragma unroll
  for (int h = 0; h < HEADS; ++h) {
    float el = __expf(logit[(size_t)i * HEADS + h] - maxb[((size_t)t * NS + d) * HEADS + h]);
    logit[(size_t)i * HEADS + h] = el;
    atomicAdd(&sumb[((size_t)t * NS + d) * HEADS + h], el);
  }
}

__global__ void k_nwp_scatter(const float* __restrict__ seq, const int* __restrict__ src,
                              const int* __restrict__ dst, const float* __restrict__ Wv,
                              const float* __restrict__ logit, const float* __restrict__ sumb,
                              float* __restrict__ msg, int E, int NN, int colbase) {
  int i = blockIdx.x * blockDim.x + threadIdx.x;
  if (i >= TT * E) return;
  int t = i / E, e = i % E;
  const float* fp = seq + ((size_t)t * NN + src[e]) * 3;
  float f0 = fp[0], f1 = fp[1], f2 = fp[2];
  int d = dst[e];
  size_t mb = ((size_t)t * NS + d) * 64 + colbase;
#pragma unroll
  for (int h = 0; h < HEADS; ++h) {
    float alpha = logit[(size_t)i * HEADS + h] /
                  (sumb[((size_t)t * NS + d) * HEADS + h] + 1e-9f);
#pragma unroll
    for (int j = 0; j < HD; ++j) {
      int c = h * HD + j;
      float val = f0 * Wv[c] + f1 * Wv[32 + c] + f2 * Wv[64 + c];  // Wv is (3,32)
      atomicAdd(&msg[mb + c], alpha * val);
    }
  }
}

// ---------------------------------------------------------------------------
extern "C" void kernel_launch(void* const* d_in, const int* in_sizes, int n_in,
                              void* d_out, int out_size, void* d_ws, size_t ws_size,
                              hipStream_t stream) {
  (void)in_sizes; (void)n_in; (void)out_size; (void)ws_size;
  const float* meas    = (const float*)d_in[0];
  const float* icond   = (const float*)d_in[1];
  const float* ecmwf   = (const float*)d_in[2];
  const float* stat    = (const float*)d_in[3];
  const int*   s2s     = (const int*)d_in[4];
  const float* s2s_w   = (const float*)d_in[5];
  const int*   i2s     = (const int*)d_in[6];
  const float* i2s_ea  = (const float*)d_in[7];
  const int*   e2s     = (const int*)d_in[8];
  const float* e2s_ea  = (const float*)d_in[9];
  const float* icon_Wf = (const float*)d_in[10];
  const float* icon_Wv = (const float*)d_in[11];
  const float* icon_We = (const float*)d_in[12];
  const float* icon_a  = (const float*)d_in[13];
  const float* ecm_Wf  = (const float*)d_in[14];
  const float* ecm_Wv  = (const float*)d_in[15];
  const float* ecm_We  = (const float*)d_in[16];
  const float* ecm_a   = (const float*)d_in[17];
  const float* nwp_Wout= (const float*)d_in[18];
  const float* nwp_bout= (const float*)d_in[19];
  const float* Wg0 = (const float*)d_in[20];
  const float* bg0 = (const float*)d_in[21];
  const float* Wc0 = (const float*)d_in[22];
  const float* bc0 = (const float*)d_in[23];
  const float* Wg1 = (const float*)d_in[24];
  const float* bg1 = (const float*)d_in[25];
  const float* Wc1 = (const float*)d_in[26];
  const float* bc1 = (const float*)d_in[27];

  // -------- workspace layout (~156 MB) --------
  float* ws = (float*)d_ws;
  size_t off = 0;
  auto alloc = [&](size_t n) { float* p = ws + off; off += n; return p; };
  float* msg   = alloc((size_t)TT * NS * 64);       // attention messages (icon|ecm)
  float* nwpm  = alloc((size_t)TT * NS * 32);       // nwp_msgs after Wout
  float* logit = alloc((size_t)TT * EI * HEADS);    // reused for both edge types
  float* maxb  = alloc((size_t)TT * NS * HEADS);
  float* sumb  = alloc((size_t)TT * NS * HEADS);
  float* dc0   = alloc((size_t)NS * 576);           // cell0 diff-conv (K padded 552->576)
  float* dc1   = alloc((size_t)NS * 768);           // cell1 diff-conv
  float* g0    = alloc((size_t)NS * 256);
  float* g1    = alloc((size_t)NS * 256);
  float* c0    = alloc((size_t)NS * 128);
  float* c1    = alloc((size_t)NS * 128);
  float* wfa_i = alloc(12); float* wea_i = alloc(12);
  float* wfa_e = alloc(12); float* wea_e = alloc(12);
  unsigned short* bw = (unsigned short*)(ws + off);   // 16B-aligned (all counts %4==0)
  unsigned short* WgB0p  = bw; bw += 576 * 256;       // packed weights (bf16)
  unsigned short* WcB0p  = bw; bw += 576 * 128;
  unsigned short* WgB1p  = bw; bw += 768 * 256;
  unsigned short* WcB1p  = bw; bw += 768 * 128;
  unsigned short* WnwpBp = bw; bw += 64 * 32;
  unsigned short* dcB0   = bw; bw += (size_t)NS * 576;     // bf16 activations
  unsigned short* dcB1   = bw; bw += (size_t)NS * 768;
  unsigned short* msgB   = bw; bw += (size_t)TT * NS * 64;

  float* h0 = (float*)d_out;                 // output lives here, updated in place
  float* h1 = h0 + (size_t)NS * HIDF;

  const int BT = 256;
  auto blocks = [](long n) { return (int)((n + 255) / 256); };

  // -------- init + weight prep --------
  k_fill<<<blocks(2 * NS * HIDF), BT, 0, stream>>>((float*)d_out, 2 * NS * HIDF, 0.f);
  k_fill<<<blocks((long)TT * NS * 64), BT, 0, stream>>>(msg, TT * NS * 64, 0.f);
  k_nwp_prep<<<1, 32, 0, stream>>>(icon_Wf, icon_We, icon_a, wfa_i, wea_i);
  k_nwp_prep<<<1, 32, 0, stream>>>(ecm_Wf,  ecm_We,  ecm_a,  wfa_e, wea_e);
  k_pack_b<<<blocks(576 * 256), BT, 0, stream>>>(Wg0, WgB0p, 552, 576, 256);
  k_pack_b<<<blocks(576 * 128), BT, 0, stream>>>(Wc0, WcB0p, 552, 576, 128);
  k_pack_b<<<blocks(768 * 256), BT, 0, stream>>>(Wg1, WgB1p, 768, 768, 256);
  k_pack_b<<<blocks(768 * 128), BT, 0, stream>>>(Wc1, WcB1p, 768, 768, 128);
  k_pack_b<<<blocks(64 * 32),   BT, 0, stream>>>(nwp_Wout, WnwpBp, 64, 64, 32);

  // -------- NWP attention (all T in parallel), icon then ecm --------
  const int* i_src = i2s; const int* i_dst = i2s + EI;
  k_fill<<<blocks((long)TT * NS * HEADS), BT, 0, stream>>>(maxb, TT * NS * HEADS, -1e30f);
  k_fill<<<blocks((long)TT * NS * HEADS), BT, 0, stream>>>(sumb, TT * NS * HEADS, 0.f);
  k_nwp_logit<<<blocks((long)TT * EI), BT, 0, stream>>>(icond, i_src, i_dst, i2s_ea,
                                                        wfa_i, wea_i, logit, maxb, EI, NI);
  k_nwp_exp<<<blocks((long)TT * EI), BT, 0, stream>>>(i_dst, logit, maxb, sumb, EI);
  k_nwp_scatter<<<blocks((long)TT * EI), BT, 0, stream>>>(icond, i_src, i_dst, icon_Wv,
                                                          logit, sumb, msg, EI, NI, 0);
  const int* e_src = e2s; const int* e_dst = e2s + EE;
  k_fill<<<blocks((long)TT * NS * HEADS), BT, 0, stream>>>(maxb, TT * NS * HEADS, -1e30f);
  k_fill<<<blocks((long)TT * NS * HEADS), BT, 0, stream>>>(sumb, TT * NS * HEADS, 0.f);
  k_nwp_logit<<<blocks((long)TT * EE), BT, 0, stream>>>(ecmwf, e_src, e_dst, e2s_ea,
                                                        wfa_e, wea_e, logit, maxb, EE, NE);
  k_nwp_exp<<<blocks((long)TT * EE), BT, 0, stream>>>(e_dst, logit, maxb, sumb, EE);
  k_nwp_scatter<<<blocks((long)TT * EE), BT, 0, stream>>>(ecmwf, e_src, e_dst, ecm_Wv,
                                                          logit, sumb, msg, EE, NE, 32);
  { // nwp_msgs = msg(192000x64) @ Wout(64x32) + b   (WMMA, NTW=2)
    k_cvt_bf16<<<blocks((long)TT * NS * 64), BT, 0, stream>>>(msg, msgB, TT * NS * 64);
    int nw = ((TT * NS) / 16) * ((32 / 16) / 2);
    k_gemm_wmma<2><<<(nw + 7) / 8, 256, 0, stream>>>(msgB, WnwpBp, nwp_bout, nwpm,
                                                     TT * NS, 32, 64, 0);
  }

  // -------- sequential DCGRU scan --------
  const int* s_src = s2s; const int* s_dst = s2s + ES;
  const int nwG = (NS / 16) * ((256 / 16) / 4);   // gate GEMM waves (NTW=4)
  const int nwC = (NS / 16) * ((128 / 16) / 4);   // candidate GEMM waves (NTW=4)

  for (int t = 0; t < TT; ++t) {
    const float* meas_t = meas + (size_t)t * NS * MEASF;
    const float* nwp_t  = nwpm + (size_t)t * NS * NWPF;

    // ---- cell 0: gates ----
    k_build_dc0<<<blocks(NS * 576), BT, 0, stream>>>(meas_t, nwp_t, stat, h0, nullptr, dc0);
    k_spmv_hop<<<blocks((long)ES * 184), BT, 0, stream>>>(s_src, s_dst, s2s_w, ES, dc0, 576, 0, 184, 184);
    k_spmv_hop<<<blocks((long)ES * 184), BT, 0, stream>>>(s_src, s_dst, s2s_w, ES, dc0, 576, 184, 368, 184);
    k_cvt_bf16<<<blocks(NS * 576), BT, 0, stream>>>(dc0, dcB0, NS * 576);
    k_gemm_wmma<4><<<(nwG + 7) / 8, 256, 0, stream>>>(dcB0, WgB0p, bg0, g0, NS, 256, 576, 1);
    // ---- cell 0: candidate (uses r*h0, old h0) ----
    k_build_dc0<<<blocks(NS * 576), BT, 0, stream>>>(meas_t, nwp_t, stat, h0, g0, dc0);
    k_spmv_hop<<<blocks((long)ES * 184), BT, 0, stream>>>(s_src, s_dst, s2s_w, ES, dc0, 576, 0, 184, 184);
    k_spmv_hop<<<blocks((long)ES * 184), BT, 0, stream>>>(s_src, s_dst, s2s_w, ES, dc0, 576, 184, 368, 184);
    k_cvt_bf16<<<blocks(NS * 576), BT, 0, stream>>>(dc0, dcB0, NS * 576);
    k_gemm_wmma<4><<<(nwC + 7) / 8, 256, 0, stream>>>(dcB0, WcB0p, bc0, c0, NS, 128, 576, 2);
    k_update<<<blocks(NS * HIDF), BT, 0, stream>>>(h0, g0, c0);

    // ---- cell 1: gates ----
    k_build_dc1<<<blocks(NS * 768), BT, 0, stream>>>(h0, h1, nullptr, dc1);
    k_spmv_hop<<<blocks((long)ES * 256), BT, 0, stream>>>(s_src, s_dst, s2s_w, ES, dc1, 768, 0, 256, 256);
    k_spmv_hop<<<blocks((long)ES * 256), BT, 0, stream>>>(s_src, s_dst, s2s_w, ES, dc1, 768, 256, 512, 256);
    k_cvt_bf16<<<blocks(NS * 768), BT, 0, stream>>>(dc1, dcB1, NS * 768);
    k_gemm_wmma<4><<<(nwG + 7) / 8, 256, 0, stream>>>(dcB1, WgB1p, bg1, g1, NS, 256, 768, 1);
    // ---- cell 1: candidate ----
    k_build_dc1<<<blocks(NS * 768), BT, 0, stream>>>(h0, h1, g1, dc1);
    k_spmv_hop<<<blocks((long)ES * 256), BT, 0, stream>>>(s_src, s_dst, s2s_w, ES, dc1, 768, 0, 256, 256);
    k_spmv_hop<<<blocks((long)ES * 256), BT, 0, stream>>>(s_src, s_dst, s2s_w, ES, dc1, 768, 256, 512, 256);
    k_cvt_bf16<<<blocks(NS * 768), BT, 0, stream>>>(dc1, dcB1, NS * 768);
    k_gemm_wmma<4><<<(nwC + 7) / 8, 256, 0, stream>>>(dcB1, WcB1p, bc1, c1, NS, 128, 768, 2);
    k_update<<<blocks(NS * HIDF), BT, 0, stream>>>(h1, g1, c1);
  }
}
